// Stage2GNN_9019431321967
// MI455X (gfx1250) — matrix operand order repaired and checked by reading, not compiled, not run on hardware
//
#include <hip/hip_runtime.h>
#include <hip/hip_bf16.h>

typedef __attribute__((ext_vector_type(16))) _Float16 v16h;
typedef __attribute__((ext_vector_type(8)))  float    v8f;

#define HDIM      128
#define SATDIM    32
#define NEG_SLOPE 0.2f
#define BN_EPS    1e-5f

// ---------------------------------------------------------------------------
// Stage 0: fused input projections + gating
//   sat = relu(x[:, :32] @ sat_w^T + sat_b)
//   nei = relu(x[:, 32:] @ neigh_w^T + neigh_b)
//   gate = sigmoid([sat,nei] @ fus_w^T + fus_b)
//   h = gate*sat + (1-gate)*nei     (also write f16 mirror for WMMA)
// ---------------------------------------------------------------------------
__global__ void input_fuse_kernel(const float* __restrict__ x,
                                  const float* __restrict__ sat_w, const float* __restrict__ sat_b,
                                  const float* __restrict__ neigh_w, const float* __restrict__ neigh_b,
                                  const float* __restrict__ fus_w, const float* __restrict__ fus_b,
                                  float* __restrict__ h, _Float16* __restrict__ h16, int N) {
    int i = blockIdx.x;
    if (i >= N) return;
    int j = threadIdx.x;                 // feature 0..127
    __shared__ float xr[2 * SATDIM];
    __shared__ float red[HDIM];
    if (j < 2 * SATDIM) xr[j] = x[(size_t)i * (2 * SATDIM) + j];
    __syncthreads();

    float s = sat_b[j];
    float n = neigh_b[j];
#pragma unroll
    for (int k = 0; k < SATDIM; ++k) {
        s += xr[k]          * sat_w[j * SATDIM + k];
        n += xr[SATDIM + k] * neigh_w[j * SATDIM + k];
    }
    s = fmaxf(s, 0.f);
    n = fmaxf(n, 0.f);

    red[j] = fus_w[j] * s + fus_w[HDIM + j] * n;
    __syncthreads();
    for (int off = HDIM / 2; off > 0; off >>= 1) {
        if (j < off) red[j] += red[j + off];
        __syncthreads();
    }
    float g = 1.f / (1.f + __expf(-(red[0] + fus_b[0])));
    float hv = g * s + (1.f - g) * n;
    size_t idx = (size_t)i * HDIM + j;
    h[idx]   = hv;
    h16[idx] = (_Float16)hv;
}

// ---------------------------------------------------------------------------
// Convert all GEMM weights to f16 once.
// layout in w16: [w1 16384][w2 16384][fc1_w 16384][fc2_w 2048]
// ---------------------------------------------------------------------------
__global__ void conv_w16_kernel(const float* __restrict__ w1, const float* __restrict__ w2,
                                const float* __restrict__ fc1w, const float* __restrict__ fc2w,
                                _Float16* __restrict__ w16) {
    const int T = 3 * HDIM * HDIM + 16 * HDIM; // 51200
    for (int idx = blockIdx.x * blockDim.x + threadIdx.x; idx < T; idx += gridDim.x * blockDim.x) {
        float v;
        if (idx < 16384)       v = w1[idx];
        else if (idx < 32768)  v = w2[idx - 16384];
        else if (idx < 49152)  v = fc1w[idx - 32768];
        else                   v = fc2w[idx - 49152];
        w16[idx] = (_Float16)v;
    }
}

// ---------------------------------------------------------------------------
// WMMA GEMM:  out = act(A[Nrows,K] @ W[Hout,K]^T + bias)
// Each wave computes a 16(M)x32(N) tile: two independent 16x16 accumulators
// sharing one A fragment (halves A traffic, two interleavable WMMA chains to
// cover the WMMA->WMMA XDL hazard).  v_wmma_f32_16x16x32_f16 over K.
// gfx1250 layouts (CDNA5 ISA):
//   A (16x32 f16): lane m=lane&15; VGPR v<4: K = 8*(lane>=16) + 2v + p
//                                  VGPR v>=4: K = 16 + 8*(lane>=16) + 2(v-4) + p
//   B (32x16 f16): lane col = lane&15; K same as A (table order)
//   C/D (16x16 f32): row = v + 8*(lane>=16), col = lane&15
// ---------------------------------------------------------------------------
__global__ void wmma_gemm_kernel(const _Float16* __restrict__ A, const _Float16* __restrict__ W,
                                 const float* __restrict__ bias,
                                 float* __restrict__ outF, _Float16* __restrict__ outH,
                                 int Nrows, int K, int Hout, int act) {
    int lane     = threadIdx.x & 31;
    int wave     = threadIdx.x >> 5;
    int wavesPB  = blockDim.x >> 5;
    int laneHalf = lane >> 4;
    int laneMod  = lane & 15;

    int m0 = blockIdx.x * 16;
    int n0 = (blockIdx.y * wavesPB + wave) * 32;   // 32 output cols per wave
    if (n0 >= Hout) return;
    bool has2 = (n0 + 16) < Hout;

    int rowA = m0 + laneMod;
    const _Float16* Arow  = A + (size_t)((rowA < Nrows) ? rowA : 0) * K;
    const _Float16* Wrow0 = W + (size_t)(n0 + laneMod) * K;
    const _Float16* Wrow1 = W + (size_t)(has2 ? (n0 + 16 + laneMod) : laneMod) * K;

    v8f acc0 = {};
    v8f acc1 = {};
    for (int k0 = 0; k0 < K; k0 += 32) {
        v16h a, b0, b1;
#pragma unroll
        for (int j = 0; j < 16; ++j) {
            int v = j >> 1, p = j & 1;
            int kk = (v < 4) ? (laneHalf * 8 + 2 * v + p)
                             : (16 + laneHalf * 8 + 2 * (v - 4) + p);
            a[j]  = Arow[k0 + kk];
            b0[j] = Wrow0[k0 + kk];
            b1[j] = Wrow1[k0 + kk];
        }
        acc0 = __builtin_amdgcn_wmma_f32_16x16x32_f16(false, a, false, b0,
                                                      (short)0, acc0, false, false);
        acc1 = __builtin_amdgcn_wmma_f32_16x16x32_f16(false, a, false, b1,
                                                      (short)0, acc1, false, false);
    }

#pragma unroll
    for (int v = 0; v < 8; ++v) {
        int row = m0 + v + 8 * laneHalf;
        if (row >= Nrows) continue;
        {
            int col = n0 + laneMod;
            float val = acc0[v] + (bias ? bias[col] : 0.f);
            if (act) val = fmaxf(val, 0.f);
            size_t o = (size_t)row * Hout + col;
            if (outF) outF[o] = val;
            if (outH) outH[o] = (_Float16)val;
        }
        if (has2) {
            int col = n0 + 16 + laneMod;
            float val = acc1[v] + (bias ? bias[col] : 0.f);
            if (act) val = fmaxf(val, 0.f);
            size_t o = (size_t)row * Hout + col;
            if (outF) outF[o] = val;
            if (outH) outH[o] = (_Float16)val;
        }
    }
}

// ---------------------------------------------------------------------------
// Per-node attention logits: al_s[i] = hW[i,:]·a_src ; al_d[i] = hW[i,:]·a_dst
// one wave per node; lane holds a float4 (128/32 = 4 elems).
// ---------------------------------------------------------------------------
__global__ void row_dot2_kernel(const float* __restrict__ hW,
                                const float* __restrict__ avs, const float* __restrict__ avd,
                                float* __restrict__ al_s, float* __restrict__ al_d, int N) {
    int gt   = blockIdx.x * blockDim.x + threadIdx.x;
    int node = gt >> 5;
    int lane = gt & 31;
    if (node >= N) return;
    float4 v  = ((const float4*)(hW + (size_t)node * HDIM))[lane];
    float4 as = ((const float4*)avs)[lane];
    float4 ad = ((const float4*)avd)[lane];
    float s = v.x * as.x + v.y * as.y + v.z * as.z + v.w * as.w;
    float d = v.x * ad.x + v.y * ad.y + v.z * ad.z + v.w * ad.w;
#pragma unroll
    for (int off = 16; off > 0; off >>= 1) {
        s += __shfl_down(s, off, 32);
        d += __shfl_down(d, off, 32);
    }
    if (lane == 0) { al_s[node] = s; al_d[node] = d; }
}

// ---------------------------------------------------------------------------
// Zero per-layer scratch: agg[N*128], m_ord[N]=0(ordered -max), ssum[N], stats.
// ---------------------------------------------------------------------------
__global__ void zero_layer_kernel(float* __restrict__ agg, unsigned* __restrict__ m_ord,
                                  float* __restrict__ ssum, float* __restrict__ sums,
                                  float* __restrict__ sumsq, int N) {
    size_t total = (size_t)N * HDIM;
    for (size_t idx = blockIdx.x * (size_t)blockDim.x + threadIdx.x; idx < total;
         idx += (size_t)gridDim.x * blockDim.x) {
        agg[idx] = 0.f;
        if (idx < (size_t)N) { m_ord[idx] = 0u; ssum[idx] = 0.f; }
        if (idx < HDIM)      { sums[idx] = 0.f; sumsq[idx] = 0.f; }
    }
}

__device__ __forceinline__ unsigned float_to_ordered(float f) {
    unsigned b = __float_as_uint(f);
    return (b & 0x80000000u) ? ~b : (b | 0x80000000u);
}
__device__ __forceinline__ float ordered_to_float(unsigned u) {
    unsigned b = (u & 0x80000000u) ? (u & 0x7FFFFFFFu) : ~u;
    return __uint_as_float(b);
}

// pass A: alpha = leaky_relu(al_s[src]+al_d[dst]); segment-max via ordered atomicMax
__global__ void edge_alpha_max_kernel(const int* __restrict__ src, const int* __restrict__ dst,
                                      const float* __restrict__ al_s, const float* __restrict__ al_d,
                                      float* __restrict__ ebuf, unsigned* __restrict__ m_ord,
                                      int E, int ET) {
    for (int e = blockIdx.x * blockDim.x + threadIdx.x; e < ET; e += gridDim.x * blockDim.x) {
        int s_, d_;
        if (e < E) { s_ = src[e]; d_ = dst[e]; } else { s_ = d_ = e - E; }
        float a = al_s[s_] + al_d[d_];
        a = (a > 0.f) ? a : NEG_SLOPE * a;
        ebuf[e] = a;
        atomicMax(&m_ord[d_], float_to_ordered(a));
    }
}

// pass B: e = exp(alpha - max[dst]); segment-sum
__global__ void edge_exp_sum_kernel(const int* __restrict__ src, const int* __restrict__ dst,
                                    const unsigned* __restrict__ m_ord,
                                    float* __restrict__ ebuf, float* __restrict__ ssum,
                                    int E, int ET) {
    for (int e = blockIdx.x * blockDim.x + threadIdx.x; e < ET; e += gridDim.x * blockDim.x) {
        int d_ = (e < E) ? dst[e] : (e - E);
        float ex = __expf(ebuf[e] - ordered_to_float(m_ord[d_]));
        ebuf[e] = ex;
        atomicAdd(&ssum[d_], ex);
    }
}

// pass C: agg[dst] += (e / s[dst]) * hW[src]; one wave32 per edge, float4 per lane
__global__ void edge_aggregate_kernel(const int* __restrict__ src, const int* __restrict__ dst,
                                      const float* __restrict__ ebuf, const float* __restrict__ ssum,
                                      const float* __restrict__ hW, float* __restrict__ agg,
                                      int E, int ET) {
    int gt   = blockIdx.x * blockDim.x + threadIdx.x;
    int lane = gt & 31;
    int nw   = (gridDim.x * blockDim.x) >> 5;
    for (int e = gt >> 5; e < ET; e += nw) {
        int s_, d_;
        if (e < E) { s_ = src[e]; d_ = dst[e]; } else { s_ = d_ = e - E; }
        float w = ebuf[e] / (ssum[d_] + 1e-16f);
        float4 v = ((const float4*)(hW + (size_t)s_ * HDIM))[lane];
        float* o = agg + (size_t)d_ * HDIM + lane * 4;
        atomicAdd(o + 0, w * v.x);
        atomicAdd(o + 1, w * v.y);
        atomicAdd(o + 2, w * v.z);
        atomicAdd(o + 3, w * v.w);
    }
}

// add GAT bias in place + per-feature partial sums for BN (64 rows per block)
__global__ void bias_stats_kernel(float* __restrict__ agg, const float* __restrict__ b,
                                  float* __restrict__ sums, float* __restrict__ sumsq, int N) {
    int j  = threadIdx.x;            // feature
    int n0 = blockIdx.x * 64;
    float bs = b[j], s = 0.f, q = 0.f;
    for (int r = 0; r < 64; ++r) {
        int n = n0 + r;
        if (n < N) {
            size_t idx = (size_t)n * HDIM + j;
            float v = agg[idx] + bs;
            agg[idx] = v;
            s += v; q += v * v;
        }
    }
    atomicAdd(&sums[j], s);
    atomicAdd(&sumsq[j], q);
}

// h = relu(BN(agg)) + h   (in-place residual; refresh f16 mirror)
__global__ void bn_relu_res_kernel(const float* __restrict__ agg,
                                   const float* __restrict__ sums, const float* __restrict__ sumsq,
                                   const float* __restrict__ g, const float* __restrict__ be,
                                   float* __restrict__ h, _Float16* __restrict__ h16,
                                   int N, float invN) {
    size_t total = (size_t)N * HDIM;
    for (size_t idx = blockIdx.x * (size_t)blockDim.x + threadIdx.x; idx < total;
         idx += (size_t)gridDim.x * blockDim.x) {
        int j = (int)(idx & (HDIM - 1));
        float mu  = sums[j] * invN;
        float var = sumsq[j] * invN - mu * mu;
        float y = (agg[idx] - mu) * rsqrtf(var + BN_EPS) * g[j] + be[j];
        y = fmaxf(y, 0.f) + h[idx];
        h[idx]   = y;
        h16[idx] = (_Float16)y;
    }
}

// ---------------------------------------------------------------------------
// Host-side launcher
// ---------------------------------------------------------------------------
extern "C" void kernel_launch(void* const* d_in, const int* in_sizes, int n_in,
                              void* d_out, int out_size, void* d_ws, size_t ws_size,
                              hipStream_t stream) {
    const float* x       = (const float*)d_in[0];
    const int*   ei      = (const int*)d_in[1];
    const float* sat_w   = (const float*)d_in[2];
    const float* sat_b   = (const float*)d_in[3];
    const float* neigh_w = (const float*)d_in[4];
    const float* neigh_b = (const float*)d_in[5];
    const float* fus_w   = (const float*)d_in[6];
    const float* fus_b   = (const float*)d_in[7];
    const float* w1      = (const float*)d_in[8];
    const float* as1     = (const float*)d_in[9];
    const float* ad1     = (const float*)d_in[10];
    const float* b1      = (const float*)d_in[11];
    const float* g1      = (const float*)d_in[12];
    const float* be1     = (const float*)d_in[13];
    const float* w2      = (const float*)d_in[14];
    const float* as2     = (const float*)d_in[15];
    const float* ad2     = (const float*)d_in[16];
    const float* b2      = (const float*)d_in[17];
    const float* g2      = (const float*)d_in[18];
    const float* be2     = (const float*)d_in[19];
    const float* fc1_w   = (const float*)d_in[20];
    const float* fc1_b   = (const float*)d_in[21];
    const float* fc2_w   = (const float*)d_in[22];
    const float* fc2_b   = (const float*)d_in[23];
    float* out = (float*)d_out;

    const int N  = in_sizes[0] / (2 * SATDIM);
    const int E  = in_sizes[1] / 2;
    const int ET = E + N;
    const int* src = ei;
    const int* dst = ei + E;

    // ---- workspace layout (16B-aligned blocks) ----
    char* ws = (char*)d_ws;
    size_t off = 0;
    float*     h    = (float*)(ws + off);      off += (size_t)N * HDIM * 4;
    float*     hW   = (float*)(ws + off);      off += (size_t)N * HDIM * 4;
    float*     agg  = (float*)(ws + off);      off += (size_t)N * HDIM * 4;
    _Float16*  h16  = (_Float16*)(ws + off);   off += (size_t)N * HDIM * 2;
    _Float16*  fc16 = (_Float16*)(ws + off);   off += (size_t)N * HDIM * 2;
    _Float16*  w16  = (_Float16*)(ws + off);   off += (3 * HDIM * HDIM + 16 * HDIM) * 2;
    float*     al_s = (float*)(ws + off);      off += (size_t)N * 4;
    float*     al_d = (float*)(ws + off);      off += (size_t)N * 4;
    unsigned*  mord = (unsigned*)(ws + off);   off += (size_t)N * 4;
    float*     ssum = (float*)(ws + off);      off += (size_t)N * 4;
    float*     sums = (float*)(ws + off);      off += HDIM * 4;
    float*     sumsq= (float*)(ws + off);      off += HDIM * 4;
    float*     ebuf = (float*)(ws + off);      off += (size_t)ET * 4;

    _Float16* w1h  = w16;
    _Float16* w2h  = w16 + HDIM * HDIM;
    _Float16* fc1h = w16 + 2 * HDIM * HDIM;
    _Float16* fc2h = w16 + 3 * HDIM * HDIM;

    const int mtiles = (N + 15) / 16;
    const float invN = 1.f / (float)N;

    // stage 0: input projections + fusion gate
    input_fuse_kernel<<<N, HDIM, 0, stream>>>(x, sat_w, sat_b, neigh_w, neigh_b,
                                              fus_w, fus_b, h, h16, N);
    conv_w16_kernel<<<200, 256, 0, stream>>>(w1, w2, fc1_w, fc2_w, w16);

    // two GAT layers
    const _Float16* Wl[2]  = {w1h, w2h};
    const float*    asl[2] = {as1, as2};
    const float*    adl[2] = {ad1, ad2};
    const float*    bl[2]  = {b1, b2};
    const float*    gl[2]  = {g1, g2};
    const float*    bel[2] = {be1, be2};

    for (int layer = 0; layer < 2; ++layer) {
        // hW = h @ W^T   (WMMA; 4 waves x 32 cols = 128 cols per block)
        wmma_gemm_kernel<<<dim3(mtiles, 1), 128, 0, stream>>>(
            h16, Wl[layer], nullptr, hW, nullptr, N, HDIM, HDIM, 0);
        // attention logits per node
        row_dot2_kernel<<<(N + 7) / 8, 256, 0, stream>>>(hW, asl[layer], adl[layer],
                                                         al_s, al_d, N);
        // reset accumulators
        zero_layer_kernel<<<4096, 256, 0, stream>>>(agg, mord, ssum, sums, sumsq, N);
        // segment softmax + aggregation over E + N self-loop edges
        edge_alpha_max_kernel<<<4096, 256, 0, stream>>>(src, dst, al_s, al_d, ebuf, mord, E, ET);
        edge_exp_sum_kernel<<<4096, 256, 0, stream>>>(src, dst, mord, ebuf, ssum, E, ET);
        edge_aggregate_kernel<<<8192, 256, 0, stream>>>(src, dst, ebuf, ssum, hW, agg, E, ET);
        // bias + BN stats, then BN->relu->residual (in place on h)
        bias_stats_kernel<<<(N + 63) / 64, HDIM, 0, stream>>>(agg, bl[layer], sums, sumsq, N);
        bn_relu_res_kernel<<<4096, 256, 0, stream>>>(agg, sums, sumsq, gl[layer], bel[layer],
                                                     h, h16, N, invN);
    }

    // fc1: relu(h @ fc1_w^T + fc1_b) -> f16 buffer
    wmma_gemm_kernel<<<dim3(mtiles, 1), 128, 0, stream>>>(
        h16, fc1h, fc1_b, nullptr, fc16, N, HDIM, HDIM, 1);
    // fc2: fc16 @ fc2_w^T + fc2_b -> d_out [N,16] f32  (single 16-col tile)
    wmma_gemm_kernel<<<dim3(mtiles, 1), 32, 0, stream>>>(
        fc16, fc2h, fc2_b, out, nullptr, N, HDIM, 16, 0);
}
